// ResidualModel_19576460935547
// MI455X (gfx1250) — matrix operand order repaired and checked by reading, not compiled, not run on hardware
//
#include <hip/hip_runtime.h>
#include <hip/hip_bf16.h>

#define INP_SIZE 3072
#define HID      1024
#define OUT_DIM  768
#define BSZ      8192
#define EPS      1e-5f

typedef __attribute__((ext_vector_type(16))) __bf16 v16bf;
typedef __attribute__((ext_vector_type(8)))  float  v8f;

union Frag16 {               // 16 bf16 = 32 bytes = 8 VGPRs (packed pairs)
    uint4 u4[2];
    v16bf v;
};

__device__ __forceinline__ unsigned short f2b(float f) {
    unsigned int u = __float_as_uint(f);
    u += 0x7FFFu + ((u >> 16) & 1u);     // round-to-nearest-even
    return (unsigned short)(u >> 16);
}

__device__ __forceinline__ float blockReduceSum256(float v, float* red) {
    int tid = threadIdx.x;
    red[tid] = v; __syncthreads();
    #pragma unroll
    for (int s = 128; s > 0; s >>= 1) {
        if (tid < s) red[tid] += red[tid + s];
        __syncthreads();
    }
    float r = red[0]; __syncthreads();
    return r;
}

// gfx1250 async global->LDS copy (ASYNCcnt-tracked, no VGPR staging)
__device__ __forceinline__ void async_load_b128(const void* gsrc, void* lds_dst) {
    unsigned int loff = (unsigned int)(unsigned long long)lds_dst; // low 32 bits = LDS offset
    asm volatile("global_load_async_to_lds_b128 %0, %1, off"
                 :: "v"(loff), "v"(gsrc) : "memory");
}
__device__ __forceinline__ void wait_async0() {
    asm volatile("s_wait_asynccnt 0x0" ::: "memory");
}

// ---------------------------------------------------------------------------
// Weight transpose + f32 -> bf16 convert:  out[N,K] = bf16(in[K,N])
// ---------------------------------------------------------------------------
__global__ __launch_bounds__(256) void conv_transpose_bf16(
    const float* __restrict__ in, unsigned short* __restrict__ out, int K, int N)
{
    int idx = blockIdx.x * 256 + threadIdx.x;
    if (idx >= K * N) return;
    int k = idx / N, n = idx % N;
    out[(size_t)n * K + k] = f2b(in[idx]);
}

// ---------------------------------------------------------------------------
// LayerNorm over C columns, output bf16 (feeds WMMA A operand)
// ---------------------------------------------------------------------------
__global__ __launch_bounds__(256) void ln_to_bf16(
    const float* __restrict__ x, const float* __restrict__ g,
    const float* __restrict__ b, unsigned short* __restrict__ out, int C)
{
    __shared__ float red[256];
    int row = blockIdx.x;
    const float* xr = x + (size_t)row * C;
    float s = 0.f, s2 = 0.f;
    for (int i = threadIdx.x; i < C; i += 256) {
        float v = xr[i]; s += v; s2 += v * v;
    }
    float S  = blockReduceSum256(s,  red);
    float S2 = blockReduceSum256(s2, red);
    float mean = S / (float)C;
    float inv  = rsqrtf(S2 / (float)C - mean * mean + EPS);
    unsigned short* orow = out + (size_t)row * C;
    for (int i = threadIdx.x; i < C; i += 256)
        orow[i] = f2b((xr[i] - mean) * inv * g[i] + b[i]);
}

// ---------------------------------------------------------------------------
// Generic bf16 WMMA GEMM:  D = act(A[M,K] @ WT[N,K]^T + bias) (+ res)
//   workgroup tile 128x128, 8 waves, 32x64 per wave (2x4 of 16x16x32 wmma)
// ---------------------------------------------------------------------------
__global__ __launch_bounds__(256) void gemm_bf16_wmma(
    const unsigned short* __restrict__ A,    // [M,K] bf16 row-major
    const unsigned short* __restrict__ WT,   // [N,K] bf16 (pre-transposed weights)
    const float* __restrict__ bias,          // [N] or null
    const float* __restrict__ res,           // [M,N] or null (added AFTER relu)
    float* __restrict__ outF,                // [M,N] or null
    unsigned short* __restrict__ outB,       // [M,N] or null
    int M, int N, int K, int relu)
{
    const int lane = threadIdx.x & 31;
    const int wave = threadIdx.x >> 5;
    const int half = lane >> 4;      // lane group 0..15 / 16..31
    const int l16  = lane & 15;
    const int bm = blockIdx.x * 128 + (wave & 3) * 32;  // wave M origin
    const int bn = blockIdx.y * 128 + (wave >> 2) * 64; // wave N origin

    v8f acc[2][4] = {};

    for (int k0 = 0; k0 < K; k0 += 32) {
        Frag16 a[2], b[4];
        #pragma unroll
        for (int t = 0; t < 2; ++t) {
            const unsigned short* pa = A + (size_t)(bm + t * 16 + l16) * K + k0 + half * 16;
            a[t].u4[0] = *(const uint4*)pa;
            a[t].u4[1] = *(const uint4*)(pa + 8);
        }
        #pragma unroll
        for (int u = 0; u < 4; ++u) {
            const unsigned short* pb = WT + (size_t)(bn + u * 16 + l16) * K + k0 + half * 16;
            b[u].u4[0] = *(const uint4*)pb;
            b[u].u4[1] = *(const uint4*)(pb + 8);
        }
        if (k0 + 32 < K) {
            __builtin_prefetch(A  + (size_t)(bm + l16) * K + k0 + 32, 0, 1);
            __builtin_prefetch(WT + (size_t)(bn + l16) * K + k0 + 32, 0, 1);
        }
        #pragma unroll
        for (int t = 0; t < 2; ++t)
            #pragma unroll
            for (int u = 0; u < 4; ++u)
                acc[t][u] = __builtin_amdgcn_wmma_f32_16x16x32_bf16(
                    false, a[t].v, false, b[u].v, (short)0, acc[t][u], false, false);
    }

    #pragma unroll
    for (int t = 0; t < 2; ++t) {
        #pragma unroll
        for (int u = 0; u < 4; ++u) {
            int n = bn + u * 16 + l16;
            float bv = bias ? bias[n] : 0.f;
            #pragma unroll
            for (int v = 0; v < 8; ++v) {
                int m = bm + t * 16 + half * 8 + v;
                float x = acc[t][u][v] + bv;
                if (relu) x = fmaxf(x, 0.f);
                size_t o = (size_t)m * N + n;
                if (res)  x += res[o];
                if (outF) outF[o] = x;
                if (outB) outB[o] = f2b(x);
            }
        }
    }
}

// ---------------------------------------------------------------------------
// LN2 + row-normalize: pred (f32, to d_out) and pn (bf16, for S GEMM)
// ---------------------------------------------------------------------------
__global__ __launch_bounds__(256) void ln2_normalize(
    const float* __restrict__ Z, const float* __restrict__ g,
    const float* __restrict__ b, float* __restrict__ pred,
    unsigned short* __restrict__ PN)
{
    __shared__ float red[256];
    const int C = OUT_DIM;
    int row = blockIdx.x;
    const float* zr = Z + (size_t)row * C;
    float lv[3];
    float s = 0.f, s2 = 0.f;
    #pragma unroll
    for (int i = 0; i < 3; ++i) {
        float v = zr[threadIdx.x + 256 * i];
        lv[i] = v; s += v; s2 += v * v;
    }
    float S  = blockReduceSum256(s,  red);
    float S2 = blockReduceSum256(s2, red);
    float mean = S / (float)C;
    float inv  = rsqrtf(S2 / (float)C - mean * mean + EPS);
    float ns = 0.f;
    #pragma unroll
    for (int i = 0; i < 3; ++i) {
        int c = threadIdx.x + 256 * i;
        float t = (lv[i] - mean) * inv * g[c] + b[c];
        lv[i] = t; ns += t * t;
    }
    float NS = blockReduceSum256(ns, red);
    float rn = rsqrtf(NS);
    #pragma unroll
    for (int i = 0; i < 3; ++i) {
        int c = threadIdx.x + 256 * i;
        pred[(size_t)row * C + c] = lv[i];
        PN[(size_t)row * C + c]   = f2b(lv[i] * rn);
    }
}

__global__ __launch_bounds__(256) void rownorm_bf16(
    const float* __restrict__ Y, unsigned short* __restrict__ YN)
{
    __shared__ float red[256];
    const int C = OUT_DIM;
    int row = blockIdx.x;
    const float* yr = Y + (size_t)row * C;
    float lv[3]; float s2 = 0.f;
    #pragma unroll
    for (int i = 0; i < 3; ++i) {
        lv[i] = yr[threadIdx.x + 256 * i];
        s2 += lv[i] * lv[i];
    }
    float NS = blockReduceSum256(s2, red);
    float rn = rsqrtf(NS);
    #pragma unroll
    for (int i = 0; i < 3; ++i)
        YN[(size_t)row * C + threadIdx.x + 256 * i] = f2b(lv[i] * rn);
}

// ---------------------------------------------------------------------------
// Streaming contrastive: S = PN @ YN^T in 32x64 WMMA chunks per wave.
//  - YN K-slabs staged into LDS with async copies (shared by all 8 waves),
//    double buffered, ASYNCcnt + barrier synced.
//  - K-loop fully unrolled (24 steps known at compile time).
//  - Online logsumexp: per-32-col segment max, single rescale per segment.
//  - argmax via rare rescan; diagonal computed directly from bf16 buffers.
// ---------------------------------------------------------------------------
__global__ __launch_bounds__(256) void contrastive_stream(
    const unsigned short* __restrict__ PN,   // [BSZ, OUT_DIM] bf16
    const unsigned short* __restrict__ YN,   // [BSZ, OUT_DIM] bf16
    float* __restrict__ loss, float* __restrict__ accp)
{
    __shared__ unsigned short Ys[2][64][32];  // double-buffered YN K-slab (8 KB)
    __shared__ float sbuf[8][32][36];         // per-wave 32x32 staging, 16B-aligned rows
    const int K = OUT_DIM;
    const int lane = threadIdx.x & 31;
    const int wave = threadIdx.x >> 5;
    const int half = lane >> 4;
    const int l16  = lane & 15;
    const int rowBase = (blockIdx.x * 8 + wave) * 32;
    const int myrow   = rowBase + lane;
    const int srow = threadIdx.x >> 2;        // async-copy mapping: 64 rows x 4 segs
    const int sseg = threadIdx.x & 3;

    float run_max = -3.402823466e38f, run_sum = 0.f;
    float amax_v  = -3.402823466e38f; int amax_i = 0;

    for (int c0 = 0; c0 < BSZ; c0 += 64) {
        __syncthreads();   // previous chunk fully consumed before restaging buf 0
        async_load_b128(YN + (size_t)(c0 + srow) * K + sseg * 8, &Ys[0][srow][sseg * 8]);

        v8f acc[2][4] = {};
        #pragma unroll
        for (int k0 = 0; k0 < K; k0 += 32) {
            int cur = (k0 >> 5) & 1;
            wait_async0();       // own slab writes landed in LDS
            __syncthreads();     // everyone's slab portion visible; prev buf free
            if (k0 + 32 < K)
                async_load_b128(YN + (size_t)(c0 + srow) * K + (k0 + 32) + sseg * 8,
                                &Ys[cur ^ 1][srow][sseg * 8]);

            Frag16 a[2], b[4];
            #pragma unroll
            for (int t = 0; t < 2; ++t) {
                const unsigned short* pa = PN + (size_t)(rowBase + t * 16 + l16) * K + k0 + half * 16;
                a[t].u4[0] = *(const uint4*)pa;
                a[t].u4[1] = *(const uint4*)(pa + 8);
            }
            #pragma unroll
            for (int u = 0; u < 4; ++u) {
                const uint4* pb = (const uint4*)&Ys[cur][u * 16 + l16][half * 16];
                b[u].u4[0] = pb[0];
                b[u].u4[1] = pb[1];
            }
            #pragma unroll
            for (int t = 0; t < 2; ++t)
                #pragma unroll
                for (int u = 0; u < 4; ++u)
                    acc[t][u] = __builtin_amdgcn_wmma_f32_16x16x32_bf16(
                        false, a[t].v, false, b[u].v, (short)0, acc[t][u], false, false);
        }

        // rotate accumulator layout through LDS; segment-wise online logsumexp
        #pragma unroll
        for (int hh = 0; hh < 2; ++hh) {
            __syncthreads();
            #pragma unroll
            for (int t = 0; t < 2; ++t)
                #pragma unroll
                for (int uu = 0; uu < 2; ++uu) {
                    int u = 2 * hh + uu;
                    #pragma unroll
                    for (int v = 0; v < 8; ++v)
                        sbuf[wave][t * 16 + half * 8 + v][uu * 16 + l16] = acc[t][u][v];
                }
            __syncthreads();

            float vals[32];
            const float4* rowp = (const float4*)sbuf[wave][lane];
            #pragma unroll
            for (int q = 0; q < 8; ++q) {
                float4 f = rowp[q];
                vals[4 * q + 0] = f.x; vals[4 * q + 1] = f.y;
                vals[4 * q + 2] = f.z; vals[4 * q + 3] = f.w;
            }
            float cmax = vals[0];
            #pragma unroll
            for (int j = 1; j < 32; ++j) cmax = fmaxf(cmax, vals[j]);
            if (cmax > amax_v) {             // rare: rescan for index
                amax_v = cmax;
                int colBase = c0 + hh * 32;
                #pragma unroll
                for (int j = 0; j < 32; ++j)
                    if (vals[j] == cmax) amax_i = colBase + j;
            }
            float nm = fmaxf(run_max, cmax);
            float ssum = 0.f;
            #pragma unroll
            for (int j = 0; j < 32; ++j) ssum += __expf(vals[j] - nm);
            run_sum = run_sum * __expf(run_max - nm) + ssum;  // one rescale per segment
            run_max = nm;
        }
    }

    // diagonal term: dot(pn_row, yn_row) directly from bf16 buffers
    float diag = 0.f;
    {
        const unsigned short* pr = PN + (size_t)myrow * K;
        const unsigned short* yr = YN + (size_t)myrow * K;
        for (int k = 0; k < K; k += 8) {
            uint4 pu = *(const uint4*)(pr + k);
            uint4 yu = *(const uint4*)(yr + k);
            const unsigned int* pw = (const unsigned int*)&pu;
            const unsigned int* yw = (const unsigned int*)&yu;
            #pragma unroll
            for (int q = 0; q < 4; ++q) {
                diag += __uint_as_float(pw[q] << 16)        * __uint_as_float(yw[q] << 16);
                diag += __uint_as_float(pw[q] & 0xFFFF0000u) * __uint_as_float(yw[q] & 0xFFFF0000u);
            }
        }
    }

    float lse = run_max + __logf(run_sum);
    atomicAdd(loss, lse - diag);                              // -sum(pn*yn) + sum(lse)
    atomicAdd(accp, (amax_i == myrow) ? (1.0f / (float)BSZ) : 0.0f);
}

// ---------------------------------------------------------------------------
extern "C" void kernel_launch(void* const* d_in, const int* in_sizes, int n_in,
                              void* d_out, int out_size, void* d_ws, size_t ws_size,
                              hipStream_t stream) {
    (void)in_sizes; (void)n_in; (void)out_size; (void)ws_size;
    const float* inp   = (const float*)d_in[0];
    const float* y     = (const float*)d_in[1];
    const float* ln1_g = (const float*)d_in[2];
    const float* ln1_b = (const float*)d_in[3];
    const float* W1    = (const float*)d_in[4];
    const float* b1    = (const float*)d_in[5];
    const float* Wa    = (const float*)d_in[6];
    const float* ba    = (const float*)d_in[7];
    const float* Wb    = (const float*)d_in[8];
    const float* bb    = (const float*)d_in[9];
    const float* W2    = (const float*)d_in[10];
    const float* b2    = (const float*)d_in[11];
    const float* ln2_g = (const float*)d_in[12];
    const float* ln2_b = (const float*)d_in[13];

    float* pred  = (float*)d_out;                       // [BSZ, OUT_DIM]
    float* lossP = pred + (size_t)BSZ * OUT_DIM;        // scalar
    float* accP  = lossP + 1;                           // scalar

    char* w = (char*)d_ws;
    float* Hf = (float*)w;              w += (size_t)BSZ * HID * 4;
    float* Zf = (float*)w;              w += (size_t)BSZ * OUT_DIM * 4;
    unsigned short* Xb  = (unsigned short*)w; w += (size_t)BSZ * INP_SIZE * 2;
    unsigned short* Hb  = (unsigned short*)w; w += (size_t)BSZ * HID * 2;
    unsigned short* Tb  = (unsigned short*)w; w += (size_t)BSZ * HID * 2;
    unsigned short* PNb = (unsigned short*)w; w += (size_t)BSZ * OUT_DIM * 2;
    unsigned short* YNb = (unsigned short*)w; w += (size_t)BSZ * OUT_DIM * 2;
    unsigned short* W1T = (unsigned short*)w; w += (size_t)INP_SIZE * HID * 2;
    unsigned short* WaT = (unsigned short*)w; w += (size_t)HID * HID * 2;
    unsigned short* WbT = (unsigned short*)w; w += (size_t)HID * HID * 2;
    unsigned short* W2T = (unsigned short*)w; w += (size_t)HID * OUT_DIM * 2;

    conv_transpose_bf16<<<(INP_SIZE * HID + 255) / 256, 256, 0, stream>>>(W1, W1T, INP_SIZE, HID);
    conv_transpose_bf16<<<(HID * HID + 255) / 256, 256, 0, stream>>>(Wa, WaT, HID, HID);
    conv_transpose_bf16<<<(HID * HID + 255) / 256, 256, 0, stream>>>(Wb, WbT, HID, HID);
    conv_transpose_bf16<<<(HID * OUT_DIM + 255) / 256, 256, 0, stream>>>(W2, W2T, HID, OUT_DIM);

    ln_to_bf16<<<BSZ, 256, 0, stream>>>(inp, ln1_g, ln1_b, Xb, INP_SIZE);

    gemm_bf16_wmma<<<dim3(BSZ / 128, HID / 128), 256, 0, stream>>>(
        Xb, W1T, b1, nullptr, Hf, Hb, BSZ, HID, INP_SIZE, 0);

    for (int r = 0; r < 2; ++r) {
        gemm_bf16_wmma<<<dim3(BSZ / 128, HID / 128), 256, 0, stream>>>(
            Hb, WaT, ba, nullptr, nullptr, Tb, BSZ, HID, HID, 1);          // t = relu(h@Wa+ba)
        gemm_bf16_wmma<<<dim3(BSZ / 128, HID / 128), 256, 0, stream>>>(
            Tb, WbT, bb, Hf, Hf, Hb, BSZ, HID, HID, 1);                    // h = h + relu(t@Wb+bb)
    }

    gemm_bf16_wmma<<<dim3(BSZ / 128, OUT_DIM / 128), 256, 0, stream>>>(
        Hb, W2T, b2, nullptr, Zf, nullptr, BSZ, OUT_DIM, HID, 0);

    ln2_normalize<<<BSZ, 256, 0, stream>>>(Zf, ln2_g, ln2_b, pred, PNb);
    rownorm_bf16<<<BSZ, 256, 0, stream>>>(y, YNb);

    hipMemsetAsync(lossP, 0, 2 * sizeof(float), stream);
    contrastive_stream<<<BSZ / 256, 256, 0, stream>>>(PNb, YNb, lossP, accP);
}
